// ContextModel_31138512896710
// MI455X (gfx1250) — compile-verified, matrix operand order checked by reading, not run on hardware
//
#include <hip/hip_runtime.h>
#include <cstdint>

// ---------------------------------------------------------------------------
// Types for CDNA5 WMMA (wave32, 16x16x32 bf16 -> f32)
// ---------------------------------------------------------------------------
typedef __bf16 bf16;
typedef __attribute__((ext_vector_type(16))) __bf16 v16bf;
typedef __attribute__((ext_vector_type(8)))  float  v8f;

#define NPIX 16384   // B*H*W = 16*32*32

__device__ __forceinline__ bf16 f2bf(float f) {
  union { float f; unsigned u; } a; a.f = f;
  unsigned r = a.u + 0x7fffu + ((a.u >> 16) & 1u);   // round-to-nearest-even
  union { unsigned short s; bf16 b; } c; c.s = (unsigned short)(r >> 16);
  return c.b;
}
__device__ __forceinline__ float eluf(float x) { return x > 0.f ? x : __expf(x) - 1.f; }
__device__ __forceinline__ float sigm(float x) { return 1.f / (1.f + __expf(-x)); }

// Build a 16x32 bf16 A/B fragment half-row: lane holds K = kb..kb+7 and kb+16..kb+23
__device__ __forceinline__ v16bf load_frag(const bf16* p) {
  v16bf r;
#pragma unroll
  for (int i = 0; i < 8; ++i) { r[i] = p[i]; r[8 + i] = p[16 + i]; }
  return r;
}

// Async global->LDS 16B copy (per-lane addresses), tracked with ASYNCcnt.
#define ASYNC_LD_B128(ldsoff, gptr)                                          \
  asm volatile("global_load_async_to_lds_b128 %0, %1, off"                   \
               :: "v"(ldsoff),                                               \
                  "v"((unsigned long long)(uintptr_t)(gptr))                 \
               : "memory")

// ---------------------------------------------------------------------------
// WMMA GEMM:  Y[N,ldY] (+)= A[16384,K](bf16) * Wt[Cop,K]^T(bf16) + bias
//  - Wt is padded to Cop = ceil128(Co) zero-filled rows -> no guards in loop
//  - double-buffered LDS tiles filled by global_load_async_to_lds_b128
// block 256 thr (8 waves); tile 128x128; wave tile 32x64 (2x4 WMMA frags)
// ---------------------------------------------------------------------------
__global__ __launch_bounds__(256) void k_gemm(
    const bf16* __restrict__ A, const bf16* __restrict__ Wt,
    const float* __restrict__ bias, float* __restrict__ Y,
    bf16* __restrict__ Ybf, int K, int Co, int ldY, int acc)
{
  __shared__ __align__(16) bf16 As[2][128 * 32];
  __shared__ __align__(16) bf16 Bs[2][128 * 32];
  const int tid = threadIdx.x;
  const int m0 = blockIdx.x * 128, n0 = blockIdx.y * 128;
  const int lane = tid & 31, wv = tid >> 5;
  const int wm = (wv >> 1) * 32, wn = (wv & 1) * 64;
  const int lo = lane & 15, hi = lane >> 4, kb = hi * 8;

  // Staging: 512 chunks of 16B per tile; thread owns rows row0/row1, 8-elem part
  const int row0 = tid >> 2;            // 0..63
  const int row1 = 64 + row0;           // 64..127
  const int part = (tid & 3) * 8;       // bf16 elems within 32-wide row

  const bf16* gA0 = A  + (size_t)(m0 + row0) * K + part;
  const bf16* gA1 = A  + (size_t)(m0 + row1) * K + part;
  const bf16* gB0 = Wt + (size_t)(n0 + row0) * K + part;
  const bf16* gB1 = Wt + (size_t)(n0 + row1) * K + part;
  const unsigned lA0 = (unsigned)(uintptr_t)&As[0][row0 * 32 + part];
  const unsigned lA1 = (unsigned)(uintptr_t)&As[0][row1 * 32 + part];
  const unsigned lB0 = (unsigned)(uintptr_t)&Bs[0][row0 * 32 + part];
  const unsigned lB1 = (unsigned)(uintptr_t)&Bs[0][row1 * 32 + part];

  v8f accv[2][4];
#pragma unroll
  for (int i = 0; i < 2; ++i)
#pragma unroll
    for (int j = 0; j < 4; ++j)
#pragma unroll
      for (int e = 0; e < 8; ++e) accv[i][j][e] = 0.f;

  const int ntile = K >> 5;
  // prologue: tile 0 -> buffer 0
  ASYNC_LD_B128(lA0, gA0);
  ASYNC_LD_B128(lA1, gA1);
  ASYNC_LD_B128(lB0, gB0);
  ASYNC_LD_B128(lB1, gB1);

  for (int t = 0; t < ntile; ++t) {
    const int buf = t & 1;
    if (t + 1 < ntile) {                       // prefetch next tile -> other buf
      const int k0 = (t + 1) << 5;
      const unsigned bo = (unsigned)((buf ^ 1) * 8192);
      ASYNC_LD_B128(lA0 + bo, gA0 + k0);
      ASYNC_LD_B128(lA1 + bo, gA1 + k0);
      ASYNC_LD_B128(lB0 + bo, gB0 + k0);
      ASYNC_LD_B128(lB1 + bo, gB1 + k0);
      asm volatile("s_wait_asynccnt 4" ::: "memory");   // tile t complete
    } else {
      asm volatile("s_wait_asynccnt 0" ::: "memory");
    }
    __syncthreads();                           // all waves' tile t visible

    v16bf af[2], bfr[4];
#pragma unroll
    for (int mi = 0; mi < 2; ++mi)
      af[mi] = load_frag(&As[buf][(wm + mi * 16 + lo) * 32 + kb]);
#pragma unroll
    for (int ni = 0; ni < 4; ++ni)
      bfr[ni] = load_frag(&Bs[buf][(wn + ni * 16 + lo) * 32 + kb]);
#pragma unroll
    for (int mi = 0; mi < 2; ++mi)
#pragma unroll
      for (int ni = 0; ni < 4; ++ni)
        accv[mi][ni] = __builtin_amdgcn_wmma_f32_16x16x32_bf16(
            false, af[mi], false, bfr[ni], (short)0, accv[mi][ni], false, false);
    __syncthreads();                           // tile t consumed by all waves
  }

#pragma unroll
  for (int mi = 0; mi < 2; ++mi)
#pragma unroll
    for (int ni = 0; ni < 4; ++ni)
#pragma unroll
      for (int v = 0; v < 8; ++v) {
        int row = m0 + wm + mi * 16 + v + hi * 8;   // D layout: M = v (+8 for hi lanes)
        int col = n0 + wn + ni * 16 + lo;
        if (col < Co) {
          float val = accv[mi][ni][v] + bias[col];
          size_t idx = (size_t)row * ldY + col;
          if (acc) val += Y[idx];
          if (Y)   Y[idx] = val;
          if (Ybf) Ybf[idx] = f2bf(val);
        }
      }
}

// ---------------------------------------------------------------------------
// Flash-style causal attention, one wave per 16-row query tile.
// keyt[b*1024+k][96], mixT[b*96+ch][1024], query[b*1024+q][96] (bf16)
// causal: j < i, plus zero-logit sink column (init m=0, s=1).
// ---------------------------------------------------------------------------
__global__ __launch_bounds__(128) void k_attn(
    const bf16* __restrict__ keyt, const bf16* __restrict__ mixT,
    const bf16* __restrict__ query, float* __restrict__ mixed)
{
  __shared__ float s_tile[4][16][33];
  __shared__ __align__(16) bf16 p_tile[4][16][32];
  __shared__ float rowsc[4][16];
  const int wv = threadIdx.x >> 5, lane = threadIdx.x & 31;
  const int lo = lane & 15, hi = lane >> 4, kb = hi * 8;
  const int gt = blockIdx.x * 4 + wv;
  const int batch = gt >> 6, qt = gt & 63;
  const int qr = qt * 16;

  const bf16* qrow = query + ((size_t)(batch * 1024 + qr + lo)) * 96;
  v16bf qa[3];
#pragma unroll
  for (int c = 0; c < 3; ++c) qa[c] = load_frag(qrow + c * 32 + kb);

  v8f O[6];
#pragma unroll
  for (int t = 0; t < 6; ++t)
#pragma unroll
    for (int e = 0; e < 8; ++e) O[t][e] = 0.f;
  float m_run = 0.f, s_run = 1.f;   // sink logit 0 folded into init

  const int nsteps = (qr + 47) >> 5;
  for (int st = 0; st < nsteps; ++st) {
    const int kbase = st * 32;
    v8f S[2];
#pragma unroll
    for (int t = 0; t < 2; ++t)
#pragma unroll
      for (int e = 0; e < 8; ++e) S[t][e] = 0.f;
#pragma unroll
    for (int nt = 0; nt < 2; ++nt) {
      const bf16* krow = keyt + ((size_t)(batch * 1024 + kbase + nt * 16 + lo)) * 96;
#pragma unroll
      for (int c = 0; c < 3; ++c) {
        v16bf bk = load_frag(krow + c * 32 + kb);
        S[nt] = __builtin_amdgcn_wmma_f32_16x16x32_bf16(
            false, qa[c], false, bk, (short)0, S[nt], false, false);
      }
    }
    // dump S (D layout) to LDS; wave-local sync only (waves have divergent loops)
#pragma unroll
    for (int nt = 0; nt < 2; ++nt)
#pragma unroll
      for (int v = 0; v < 8; ++v)
        s_tile[wv][v + hi * 8][nt * 16 + lo] = S[nt][v];
    asm volatile("s_wait_dscnt 0" ::: "memory");

    // online softmax: lane handles row=lo, half hi (cols hi*16..+15)
    float vals[16];
    float mx = -1e30f;
#pragma unroll
    for (int j = 0; j < 16; ++j) {
      int col = hi * 16 + j;
      bool ok = (kbase + col) < (qr + lo);   // strict j < i
      float s = ok ? s_tile[wv][lo][col] : -1e30f;
      vals[j] = s;
      mx = fmaxf(mx, s);
    }
    mx = fmaxf(mx, __shfl_xor(mx, 16, 32));
    float nm = fmaxf(m_run, mx);
    float corr = __expf(m_run - nm);
    float ssum = 0.f;
#pragma unroll
    for (int j = 0; j < 16; ++j) {
      float p = (vals[j] > -1e29f) ? __expf(vals[j] - nm) : 0.f;
      p_tile[wv][lo][hi * 16 + j] = f2bf(p);
      ssum += p;
    }
    ssum += __shfl_xor(ssum, 16, 32);
    s_run = s_run * corr + ssum;
    m_run = nm;
    if (hi == 0) rowsc[wv][lo] = corr;
    asm volatile("s_wait_dscnt 0" ::: "memory");
#pragma unroll
    for (int v = 0; v < 8; ++v) {
      float f = rowsc[wv][v + hi * 8];
#pragma unroll
      for (int t = 0; t < 6; ++t) O[t][v] *= f;
    }
    v16bf pa = load_frag(&p_tile[wv][lo][kb]);
#pragma unroll
    for (int ct = 0; ct < 6; ++ct) {
      const bf16* mrow = mixT + ((size_t)batch * 96 + ct * 16 + lo) * 1024 + kbase + kb;
      v16bf bm = load_frag(mrow);
      O[ct] = __builtin_amdgcn_wmma_f32_16x16x32_bf16(
          false, pa, false, bm, (short)0, O[ct], false, false);
    }
  }
  if (hi == 0) rowsc[wv][lo] = s_run;
  asm volatile("s_wait_dscnt 0" ::: "memory");
#pragma unroll
  for (int v = 0; v < 8; ++v) {
    int row = qr + v + hi * 8;
    float inv = 1.f / rowsc[wv][v + hi * 8];
#pragma unroll
    for (int ct = 0; ct < 6; ++ct)
      mixed[((size_t)(batch * 1024 + row)) * 96 + ct * 16 + lo] = O[ct][v] * inv;
  }
}

// ---------------------------------------------------------------------------
// Elementwise / pack kernels
// ---------------------------------------------------------------------------
__global__ void k_cvt_wT(const float* __restrict__ W, bf16* __restrict__ dst,
                         int K, int Co, int Kp, int total) {
  int i = blockIdx.x * 256 + threadIdx.x;
  if (i >= total) return;                      // total = Cop*Kp (row-padded)
  int co = i / Kp, k = i - co * Kp;
  dst[i] = (k < K && co < Co) ? f2bf(W[(size_t)k * Co + co]) : f2bf(0.f);
}

__global__ void k_cvt_conv0(const float* __restrict__ W, bf16* __restrict__ dst) {
  int i = blockIdx.x * 256 + threadIdx.x;   // [256 rows][9*224], rows>=192 zero
  if (i >= 256 * 2016) return;
  int co = i / 2016, k = i - co * 2016;
  int t = k / 224, c = k - t * 224;
  float v = 0.f;
  if (c < 193 && co < 192) v = W[((size_t)t * 193 + c) * 192 + co];  // [3,3,193,192] HWIO
  dst[i] = f2bf(v);
}

__global__ void k_im2col(const float* __restrict__ x, bf16* __restrict__ dst) {
  int i = blockIdx.x * 256 + threadIdx.x;   // [N][9*224]
  if (i >= NPIX * 2016) return;
  int pos = i / 2016, k = i - pos * 2016;
  int t = k / 224, c = k - t * 224;
  int w = pos & 31, h = (pos >> 5) & 31, b = pos >> 10;
  int hs = h + t / 3 - 1, ws2 = w + t % 3 - 1;
  float v = 0.f;
  if (hs >= 0 && hs < 32 && ws2 >= 0 && ws2 < 32 && c < 193) {
    int sp = (b << 10) + (hs << 5) + ws2;
    v = (c < 192) ? x[(size_t)sp * 192 + c] : 1.f;   // ones channel
  }
  dst[i] = f2bf(v);
}

__device__ __forceinline__ bool shift_src(int g, int h, int w, int& hs, int& ws2) {
  hs = h; ws2 = w;
  if (g == 0)      { ws2 = w - 1; }             // right_shift
  else if (g == 1) { hs = h - 1; ws2 = w - 1; } // right+down
  else if (g == 2) { hs = h - 1; }              // down
  else             { hs = h - 1; ws2 = w + 1; } // down+left
  return hs >= 0 && hs < 32 && ws2 >= 0 && ws2 < 32;
}

__global__ void k_shift4(const float* __restrict__ src, bf16* __restrict__ dst) {
  int i = blockIdx.x * 256 + threadIdx.x;   // csn0 input: [N][192], groups of 48
  if (i >= NPIX * 192) return;
  int pos = i / 192, c = i - pos * 192;
  int w = pos & 31, h = (pos >> 5) & 31, b = pos >> 10;
  int hs, ws2;
  bool ok = shift_src(c / 48, h, w, hs, ws2);
  float v = ok ? src[((size_t)((b << 10) + (hs << 5) + ws2)) * 192 + c] : 0.f;
  dst[i] = f2bf(v);
}

__global__ void k_ce_shift4(const float* __restrict__ src, bf16* __restrict__ dst) {
  int i = blockIdx.x * 256 + threadIdx.x;   // concat_elu(ul) then shift: [N][384]
  if (i >= NPIX * 384) return;
  int pos = i / 384, c = i - pos * 384;
  int w = pos & 31, h = (pos >> 5) & 31, b = pos >> 10;
  int hs, ws2;
  bool ok = shift_src(c / 96, h, w, hs, ws2);
  float v = 0.f;
  if (ok) {
    float s = src[((size_t)((b << 10) + (hs << 5) + ws2)) * 192 + (c % 192)];
    v = eluf(c < 192 ? s : -s);
  }
  dst[i] = f2bf(v);
}

__global__ void k_concat_elu(const float* __restrict__ src, int ldS, int Cin,
                             bf16* __restrict__ dst, int Cpad) {
  int i = blockIdx.x * 256 + threadIdx.x;
  if (i >= NPIX * Cpad) return;
  int pos = i / Cpad, c = i - pos * Cpad;
  float v = 0.f;
  if (c < 2 * Cin) {
    float s = src[(size_t)pos * ldS + (c % Cin)];
    v = eluf(c < Cin ? s : -s);
  }
  dst[i] = f2bf(v);
}

__device__ __forceinline__ float content_src(const float* x, const float* ul,
                                             int pos, int cc) {
  if (cc < 192) return x[(size_t)pos * 192 + cc];
  if (cc < 384) return ul[(size_t)pos * 192 + cc - 192];
  int w = pos & 31, h = (pos >> 5) & 31;
  return cc == 384 ? (h - 16.f) / 32.f : (w - 16.f) / 32.f;
}

__global__ void k_ce_content(const float* __restrict__ x, const float* __restrict__ ul,
                             bf16* __restrict__ dst) {
  int i = blockIdx.x * 256 + threadIdx.x;   // [N][800], Cin=386
  if (i >= NPIX * 800) return;
  int pos = i / 800, c = i - pos * 800;
  float v = 0.f;
  if (c < 772) {
    float s = content_src(x, ul, pos, c % 386);
    v = eluf(c < 386 ? s : -s);
  }
  dst[i] = f2bf(v);
}

__device__ __forceinline__ float q_src(const float* ul, int pos, int cc) {
  if (cc < 192) return ul[(size_t)pos * 192 + cc];
  int w = pos & 31, h = (pos >> 5) & 31;
  return cc == 192 ? (h - 16.f) / 32.f : (w - 16.f) / 32.f;
}

__global__ void k_ce_q(const float* __restrict__ ul, bf16* __restrict__ dst) {
  int i = blockIdx.x * 256 + threadIdx.x;   // [N][416], Cin=194
  if (i >= NPIX * 416) return;
  int pos = i / 416, c = i - pos * 416;
  float v = 0.f;
  if (c < 388) {
    float s = q_src(ul, pos, c % 194);
    v = eluf(c < 194 ? s : -s);
  }
  dst[i] = f2bf(v);
}

__global__ void k_gated_add(float* __restrict__ ul, const float* __restrict__ c2, int ld) {
  int i = blockIdx.x * 256 + threadIdx.x;
  if (i >= NPIX * 192) return;
  int pos = i / 192, c = i - pos * 192;
  float u = c2[(size_t)pos * ld + c];
  float g = c2[(size_t)pos * ld + 192 + c];
  ul[i] += u * sigm(g);
}

__global__ void k_gated_add_content(const float* __restrict__ x, const float* __restrict__ ul,
                                    const float* __restrict__ c2, float* __restrict__ out) {
  int i = blockIdx.x * 256 + threadIdx.x;
  if (i >= NPIX * 386) return;
  int pos = i / 386, c = i - pos * 386;
  float u = c2[(size_t)pos * 800 + c];
  float g = c2[(size_t)pos * 800 + 386 + c];
  out[(size_t)pos * 416 + c] = content_src(x, ul, pos, c) + u * sigm(g);
}

__global__ void k_gated_add_q(const float* __restrict__ ul, const float* __restrict__ c2,
                              float* __restrict__ out) {
  int i = blockIdx.x * 256 + threadIdx.x;
  if (i >= NPIX * 194) return;
  int pos = i / 194, c = i - pos * 194;
  float u = c2[(size_t)pos * 416 + c];
  float g = c2[(size_t)pos * 416 + 194 + c];
  out[(size_t)pos * 224 + c] = q_src(ul, pos, c) + u * sigm(g);
}

__global__ void k_cvt(const float* __restrict__ src, int ldS, int Cin,
                      bf16* __restrict__ dst, int Cpad) {
  int i = blockIdx.x * 256 + threadIdx.x;
  if (i >= NPIX * Cpad) return;
  int pos = i / Cpad, c = i - pos * Cpad;
  dst[i] = f2bf(c < Cin ? src[(size_t)pos * ldS + c] : 0.f);
}

__global__ void k_elu_cvt(const float* __restrict__ src, bf16* __restrict__ dst) {
  int i = blockIdx.x * 256 + threadIdx.x;
  if (i >= NPIX * 192) return;
  dst[i] = f2bf(eluf(src[i]));
}

__global__ void k_split_km(const float* __restrict__ km, bf16* __restrict__ keyt,
                           bf16* __restrict__ mixT) {
  int i = blockIdx.x * 256 + threadIdx.x;
  if (i >= NPIX * 96) return;
  int pos = i / 96, c = i - pos * 96;
  keyt[i] = f2bf(km[(size_t)pos * 192 + c]);
  int batch = pos >> 10, key = pos & 1023;
  mixT[((size_t)batch * 96 + c) * 1024 + key] = f2bf(km[(size_t)pos * 192 + 96 + c]);
}

// ---------------------------------------------------------------------------
// Host driver
// Input order: d_in[0]=x; params flattened in JAX pytree (sorted-key) order:
//  blocks[0..3]{content(c1_W,c1_b,c2_W,c2_b), km_W, km_b,
//               out(a_W,a_b,c1_W,c1_b,c2_W,c2_b), q_W, q_b,
//               query_gr(c1_W,c1_b,c2_W,c2_b), resnets[0..1](c1_W,c1_b,c2_W,c2_b)}
//  conv0_W(105), conv0_b(106), csn0_W(107), csn0_b(108), final_W(109), final_b(110)
// ---------------------------------------------------------------------------
extern "C" void kernel_launch(void* const* d_in, const int* in_sizes, int n_in,
                              void* d_out, int out_size, void* d_ws, size_t ws_size,
                              hipStream_t stream)
{
  (void)in_sizes; (void)n_in; (void)out_size;
  if (ws_size < (size_t)280 * 1024 * 1024) return;  // arena guard
  const float* x = (const float*)d_in[0];

  char* basep = (char*)d_ws;
  size_t off = 0;
  auto alloc = [&](size_t bytes) -> void* {
    void* p = basep + off;
    off += (bytes + 255) & ~(size_t)255;
    return p;
  };

  bf16*  A_buf   = (bf16*) alloc((size_t)NPIX * 2016 * sizeof(bf16));
  float* Y_buf   = (float*)alloc((size_t)NPIX * 800 * sizeof(float));
  float* ul      = (float*)alloc((size_t)NPIX * 192 * sizeof(float));
  float* c1b     = (float*)alloc((size_t)NPIX * 416 * sizeof(float));
  float* content = (float*)alloc((size_t)NPIX * 416 * sizeof(float));
  float* qf      = (float*)alloc((size_t)NPIX * 224 * sizeof(float));
  bf16*  keyt    = (bf16*) alloc((size_t)NPIX * 96 * sizeof(bf16));
  bf16*  mixT    = (bf16*) alloc((size_t)NPIX * 96 * sizeof(bf16));
  bf16*  queryb  = (bf16*) alloc((size_t)NPIX * 96 * sizeof(bf16));
  float* mixed   = (float*)alloc((size_t)NPIX * 96 * sizeof(float));

  auto cvtw = [&](int idx, int K, int Co) -> bf16* {
    int Kp  = (K + 31) & ~31;
    int Cop = (Co + 127) & ~127;                 // row-pad so GEMM loop is guard-free
    bf16* p = (bf16*)alloc((size_t)Cop * Kp * sizeof(bf16));
    int tot = Cop * Kp;
    k_cvt_wT<<<dim3((tot + 255) / 256), dim3(256), 0, stream>>>(
        (const float*)d_in[idx], p, K, Co, Kp, tot);
    return p;
  };
  auto bias = [&](int idx) -> const float* { return (const float*)d_in[idx]; };
  auto gemm = [&](const bf16* A, const bf16* Wt, const float* bi,
                  float* Y, bf16* Yb, int K, int Co, int ldY, int acc) {
    dim3 g((NPIX + 127) / 128, (Co + 127) / 128);
    k_gemm<<<g, dim3(256), 0, stream>>>(A, Wt, bi, Y, Yb, K, Co, ldY, acc);
  };
  auto ew = [](int total) { return dim3((total + 255) / 256); };

  // ---- conv0 (3x3 'SAME' as im2col GEMM, K = 9*224) -> h0 in Y_buf ----
  bf16* Wt_conv0 = (bf16*)alloc((size_t)256 * 2016 * sizeof(bf16));
  k_cvt_conv0<<<ew(256 * 2016), dim3(256), 0, stream>>>((const float*)d_in[105], Wt_conv0);
  k_im2col<<<ew(NPIX * 2016), dim3(256), 0, stream>>>(x, A_buf);
  gemm(A_buf, Wt_conv0, bias(106), Y_buf, nullptr, 2016, 192, 192, 0);

  // ---- ul = causal_shift_nin(h0) ----
  bf16* Wt_csn0 = cvtw(107, 192, 192);
  k_shift4<<<ew(NPIX * 192), dim3(256), 0, stream>>>(Y_buf, A_buf);
  gemm(A_buf, Wt_csn0, bias(108), ul, nullptr, 192, 192, 192, 0);

  for (int b = 0; b < 4; ++b) {
    const int pb = 1 + 26 * b;
    bf16* ct_c1 = cvtw(pb + 0, 772, 386);
    bf16* ct_c2 = cvtw(pb + 2, 772, 772);
    bf16* km_W  = cvtw(pb + 4, 386, 192);
    bf16* out_a = cvtw(pb + 6, 192, 192);
    bf16* out_c1= cvtw(pb + 8, 384, 192);
    bf16* out_c2= cvtw(pb + 10, 384, 384);
    bf16* q_W   = cvtw(pb + 12, 194, 96);
    bf16* qg_c1 = cvtw(pb + 14, 388, 194);
    bf16* qg_c2 = cvtw(pb + 16, 388, 388);
    bf16 *rs_c1[2], *rs_c2[2];
    for (int r = 0; r < 2; ++r) {
      rs_c1[r] = cvtw(pb + 18 + 4 * r, 384, 192);
      rs_c2[r] = cvtw(pb + 20 + 4 * r, 384, 384);
    }

    // -- 2x gated resnet with causal_shift_nin --
    for (int r = 0; r < 2; ++r) {
      k_ce_shift4<<<ew(NPIX * 384), dim3(256), 0, stream>>>(ul, A_buf);
      gemm(A_buf, rs_c1[r], bias(pb + 19 + 4 * r), c1b, nullptr, 384, 192, 192, 0);
      k_concat_elu<<<ew(NPIX * 384), dim3(256), 0, stream>>>(c1b, 192, 192, A_buf, 384);
      gemm(A_buf, rs_c2[r], bias(pb + 21 + 4 * r), Y_buf, nullptr, 384, 384, 384, 0);
      k_gated_add<<<ew(NPIX * 192), dim3(256), 0, stream>>>(ul, Y_buf, 384);
    }

    // -- content = gated_resnet(concat(x, ul, bg)) --
    k_ce_content<<<ew(NPIX * 800), dim3(256), 0, stream>>>(x, ul, A_buf);
    gemm(A_buf, ct_c1, bias(pb + 1), c1b, nullptr, 800, 386, 416, 0);
    k_concat_elu<<<ew(NPIX * 800), dim3(256), 0, stream>>>(c1b, 416, 386, A_buf, 800);
    gemm(A_buf, ct_c2, bias(pb + 3), Y_buf, nullptr, 800, 772, 800, 0);
    k_gated_add_content<<<ew(NPIX * 386), dim3(256), 0, stream>>>(x, ul, Y_buf, content);

    // -- km -> keyt, mixin^T --
    k_cvt<<<ew(NPIX * 416), dim3(256), 0, stream>>>(content, 416, 386, A_buf, 416);
    gemm(A_buf, km_W, bias(pb + 5), Y_buf, nullptr, 416, 192, 192, 0);
    k_split_km<<<ew(NPIX * 96), dim3(256), 0, stream>>>(Y_buf, keyt, mixT);

    // -- query = nin(gated_resnet(concat(ul, bg))) --
    k_ce_q<<<ew(NPIX * 416), dim3(256), 0, stream>>>(ul, A_buf);
    gemm(A_buf, qg_c1, bias(pb + 15), c1b, nullptr, 416, 194, 224, 0);
    k_concat_elu<<<ew(NPIX * 416), dim3(256), 0, stream>>>(c1b, 224, 194, A_buf, 416);
    gemm(A_buf, qg_c2, bias(pb + 17), Y_buf, nullptr, 416, 388, 416, 0);
    k_gated_add_q<<<ew(NPIX * 194), dim3(256), 0, stream>>>(ul, Y_buf, qf);
    k_cvt<<<ew(NPIX * 224), dim3(256), 0, stream>>>(qf, 224, 194, A_buf, 224);
    gemm(A_buf, q_W, bias(pb + 13), nullptr, queryb, 224, 96, 96, 0);

    // -- causal attention --
    k_attn<<<dim3(256), dim3(128), 0, stream>>>(keyt, mixT, queryb, mixed);

    // -- out gated resnet with aux = mixed --
    k_concat_elu<<<ew(NPIX * 384), dim3(256), 0, stream>>>(ul, 192, 192, A_buf, 384);
    gemm(A_buf, out_c1, bias(pb + 9), c1b, nullptr, 384, 192, 192, 0);
    k_concat_elu<<<ew(NPIX * 192), dim3(256), 0, stream>>>(mixed, 96, 96, A_buf, 192);
    gemm(A_buf, out_a, bias(pb + 7), c1b, nullptr, 192, 192, 192, 1);   // accumulate
    k_concat_elu<<<ew(NPIX * 384), dim3(256), 0, stream>>>(c1b, 192, 192, A_buf, 384);
    gemm(A_buf, out_c2, bias(pb + 11), Y_buf, nullptr, 384, 384, 384, 0);
    k_gated_add<<<ew(NPIX * 192), dim3(256), 0, stream>>>(ul, Y_buf, 384);
  }

  // ---- final: nin(elu(ul)) -> d_out [N,384] f32 ----
  bf16* Wt_final = cvtw(109, 192, 384);
  k_elu_cvt<<<ew(NPIX * 192), dim3(256), 0, stream>>>(ul, A_buf);
  gemm(A_buf, Wt_final, bias(110), (float*)d_out, nullptr, 192, 384, 384, 0);
}
